// AOSA_76733885710837
// MI455X (gfx1250) — matrix operand order repaired and checked by reading, not compiled, not run on hardware
//
#include <hip/hip_runtime.h>
#include <hip/hip_bf16.h>
#include <math.h>

// ---------------------------------------------------------------------------
// Problem dims (B, C, N) = (8, 256, 2048)
// ---------------------------------------------------------------------------
#define BB 8
#define CC 256
#define NN 2048
#define BN (BB * NN)            // 16384
#define BN_EPS 1e-5f

typedef __attribute__((ext_vector_type(16))) __bf16 v16bf;
typedef __attribute__((ext_vector_type(8)))  float  v8f;
typedef __attribute__((ext_vector_type(8)))  unsigned short us8;
typedef unsigned short u16;

// ---------------------------------------------------------------------------
// Helpers
// ---------------------------------------------------------------------------
__device__ __forceinline__ u16 f2bf(float f) {
    union { float f; unsigned int u; } x; x.f = f;
    unsigned int u = x.u;
    unsigned int r = (u + 0x7FFFu + ((u >> 16) & 1u)) >> 16;   // RNE
    return (u16)r;
}

// A-fragment (16x32 bf16, M x K), source row-major [M][K] with stride ldK.
// Per CDNA5 ISA: lane L holds row M=L%16; element e<8 -> K=k0+half*8+e,
// e>=8 -> K=k0+16+half*8+(e-8). Both halves are contiguous 16B runs.
__device__ __forceinline__ v16bf load_frag_A(const u16* base, int row, int ldK,
                                             int k0, int half) {
    const u16* p = base + (size_t)row * ldK + k0 + half * 8;
    union { us8 u[2]; v16bf v; } r;
    r.u[0] = *(const us8*)(p);
    r.u[1] = *(const us8*)(p + 16);
    return r.v;
}

// B-fragment (32x16 bf16, K x N), source stored transposed (BT, [N][K],
// stride ldK). Lane L holds column N=L%16; element e<8 -> K=k0+half*8+e,
// e>=8 -> K=k0+16+half*8+(e-8). Both halves are contiguous 16B runs.
__device__ __forceinline__ v16bf load_frag_B(const u16* baseT, int col, int ldK,
                                             int k0, int half) {
    const u16* p = baseT + (size_t)col * ldK + k0 + half * 8;
    union { us8 u[2]; v16bf v; } r;
    r.u[0] = *(const us8*)(p);
    r.u[1] = *(const us8*)(p + 16);
    return r.v;
}

__device__ __forceinline__ v8f wmma_bf16(v16bf a, v16bf b, v8f c) {
    return __builtin_amdgcn_wmma_f32_16x16x32_bf16(
        /*neg_a=*/false, a, /*neg_b=*/false, b,
        /*c_mod=*/(short)0, c, /*reuse_a=*/false, /*reuse_b=*/false);
}

// ---------------------------------------------------------------------------
// Kernel 0: convert x -> bf16 [B][N][C] (transposed), weights -> bf16,
//           zero the atomic accumulators (colsum, bn sums).
// ---------------------------------------------------------------------------
__global__ void k0_convert(const float* __restrict__ x,
                           const float* __restrict__ Wq,
                           const float* __restrict__ Wk,
                           const float* __restrict__ Wv,
                           const float* __restrict__ Wt,
                           u16* __restrict__ xT,
                           u16* __restrict__ Wq_bf, u16* __restrict__ Wk_bf,
                           u16* __restrict__ Wv_bf, u16* __restrict__ Wt_bf,
                           float* __restrict__ colsum,
                           float* __restrict__ bnsum,
                           float* __restrict__ bnsumsq) {
    int idx = blockIdx.x * 256 + threadIdx.x;              // 0 .. B*C*N-1
    int b = idx / (CC * NN);
    int rem = idx - b * (CC * NN);
    int c = rem / NN;
    int n = rem - c * NN;
    xT[((size_t)b * NN + n) * CC + c] = f2bf(x[idx]);
    if (idx < CC * CC) {
        Wq_bf[idx] = f2bf(Wq[idx]);
        Wk_bf[idx] = f2bf(Wk[idx]);
        Wv_bf[idx] = f2bf(Wv[idx]);
        Wt_bf[idx] = f2bf(Wt[idx]);
    }
    if (idx < BN) colsum[idx] = 0.0f;
    if (idx < CC) { bnsum[idx] = 0.0f; bnsumsq[idx] = 0.0f; }
}

// ---------------------------------------------------------------------------
// Kernel 1: projections, 4-wide register blocking along the (b,n) dim.
// out[o, (b,n)] = sum_c W[o,c] * x[b,c,n]
//   proj 0 (q): -> xq [B][N][C] bf16
//   proj 1 (k): -> xk [B][N][C] bf16
//   proj 2 (v): -> xv [B][C][N] bf16, + bv
// 4096 waves per proj (16 o-tiles x 256 j-groups of 64); grid=(512,3),
// block=256 (8 waves). A fragment reused across 4 B fragments.
// ---------------------------------------------------------------------------
__global__ void k1_proj(const u16* __restrict__ Wq_bf,
                        const u16* __restrict__ Wk_bf,
                        const u16* __restrict__ Wv_bf,
                        const u16* __restrict__ xT,       // [BN][C] (BT layout)
                        const float* __restrict__ bv,
                        u16* __restrict__ xq,             // [B][N][C]
                        u16* __restrict__ xk,             // [B][N][C]
                        u16* __restrict__ xv) {           // [B][C][N]
    int lane = threadIdx.x & 31, wave = threadIdx.x >> 5;
    int half = lane >> 4, l16 = lane & 15;
    int proj = blockIdx.y;
    const u16* W = (proj == 0) ? Wq_bf : (proj == 1) ? Wk_bf : Wv_bf;

    int gtile = blockIdx.x * 8 + wave;                    // 0 .. 4095
    int o0 = (gtile >> 8) << 4;                           // channel tile
    int jg = (gtile & 255) << 6;                          // (b,n) group of 64

    v8f acc[4] = {};
    #pragma unroll
    for (int kc = 0; kc < 8; ++kc) {
        v16bf a = load_frag_A(W, o0 + l16, CC, kc * 32, half);
        #pragma unroll
        for (int u = 0; u < 4; ++u) {
            v16bf bm = load_frag_B(xT, jg + u * 16 + l16, CC, kc * 32, half);
            acc[u] = wmma_bf16(a, bm, acc[u]);
        }
    }
    #pragma unroll
    for (int u = 0; u < 4; ++u) {
        int j = jg + u * 16 + l16;                        // flat (b,n)
        int b = j >> 11, n = j & (NN - 1);
        #pragma unroll
        for (int v = 0; v < 8; ++v) {
            int o = o0 + v + half * 8;
            float val = acc[u][v];
            if (proj == 0)      xq[(size_t)j * CC + o] = f2bf(val);
            else if (proj == 1) xk[(size_t)j * CC + o] = f2bf(val);
            else xv[((size_t)b * CC + o) * NN + n] = f2bf(val + bv[o]);
        }
    }
}

// ---------------------------------------------------------------------------
// Kernel 2: fused energy GEMM + row softmax + column-sum accumulation.
// One block computes a 16-row (n) x 2048-col (m) strip of
// energy[b,n,m] = sum_c xq[b,n,c]*xk[b,m,c], softmaxes rows in a 128KB LDS
// strip, writes bf16 att^T[b][m][n], and atomically accumulates colsum[b][m].
// grid = B * N/16 = 1024 blocks, block = 256 (8 waves).
// A fragments hoisted once per block (reused across all 128 m-tiles).
// ---------------------------------------------------------------------------
__global__ void k2_energy_softmax(const u16* __restrict__ xq,   // [B][N][C]
                                  const u16* __restrict__ xk,   // [B][N][C]
                                  u16* __restrict__ attT,       // [B][m][n]
                                  float* __restrict__ colsum) { // [B][N]
    __shared__ float strip[16 * NN];                            // 128 KB
    __shared__ float rowmax[16];
    __shared__ float rowinv[16];

    int blk = blockIdx.x;
    int b = blk >> 7;
    int n0 = (blk & 127) << 4;
    const u16* A  = xq + (size_t)b * NN * CC;
    const u16* Bt = xk + (size_t)b * NN * CC;
    int lane = threadIdx.x & 31, wave = threadIdx.x >> 5;
    int half = lane >> 4, l16 = lane & 15;

    // Hoist A fragments for all 8 K-chunks (shared by every m-tile).
    v16bf afrag[8];
    #pragma unroll
    for (int kc = 0; kc < 8; ++kc)
        afrag[kc] = load_frag_A(A, n0 + l16, CC, kc * 32, half);

    // Each wave handles 16 of the 128 m-tiles.
    for (int mt = wave; mt < 128; mt += 8) {
        int m0 = mt << 4;
        v8f acc = {};
        #pragma unroll
        for (int kc = 0; kc < 8; ++kc) {
            v16bf bfrag = load_frag_B(Bt, m0 + l16, CC, kc * 32, half);
            acc = wmma_bf16(afrag[kc], bfrag, acc);
        }
        #pragma unroll
        for (int v = 0; v < 8; ++v) {
            int r = v + half * 8;
            strip[r * NN + m0 + l16] = acc[v];
        }
    }
    __syncthreads();

    // Row max + exp-sum: wave w reduces rows 2w and 2w+1 across 32 lanes.
    #pragma unroll
    for (int rr = 0; rr < 2; ++rr) {
        int r = wave * 2 + rr;
        float mx = -1e30f;
        for (int m = lane; m < NN; m += 32) mx = fmaxf(mx, strip[r * NN + m]);
        #pragma unroll
        for (int off = 16; off >= 1; off >>= 1) mx = fmaxf(mx, __shfl_xor(mx, off));
        float s = 0.0f;
        for (int m = lane; m < NN; m += 32) s += __expf(strip[r * NN + m] - mx);
        #pragma unroll
        for (int off = 16; off >= 1; off >>= 1) s += __shfl_xor(s, off);
        if (lane == 0) { rowmax[r] = mx; rowinv[r] = 1.0f / s; }
    }
    __syncthreads();

    // Column pass: normalize, emit bf16 att^T, accumulate column sums.
    for (int m = threadIdx.x; m < NN; m += 256) {
        float cs = 0.0f;
        u16* outp = attT + (size_t)b * NN * NN + (size_t)m * NN + n0;
        #pragma unroll
        for (int r = 0; r < 16; ++r) {
            float p = __expf(strip[r * NN + m] - rowmax[r]) * rowinv[r];
            cs += p;
            outp[r] = f2bf(p);
        }
        atomicAdd(&colsum[b * NN + m], cs);
    }
}

// ---------------------------------------------------------------------------
// Kernel 3: apply attention + residual diff, 4-wide register blocking (m).
// x_r[b,c,m] = (sum_n xv[b,c,n]*att[b,n,m]) / (1e-9 + colsum[b,m])
// y = x - x_r, stored bf16 transposed [B][N][C] (BT layout for kernel 4).
// 4096 waves (16 c-tiles x 32 m-groups of 64 per batch); grid=512, block=256.
// A fragment reused across 4 B fragments per K-chunk.
// ---------------------------------------------------------------------------
__global__ void k3_apply(const u16* __restrict__ xv,      // [B][C][N]
                         const u16* __restrict__ attT,    // [B][m][n]
                         const float* __restrict__ colsum,
                         const float* __restrict__ x,     // [B][C][N] f32
                         u16* __restrict__ yT) {          // [B][N][C]
    int lane = threadIdx.x & 31, wave = threadIdx.x >> 5;
    int half = lane >> 4, l16 = lane & 15;
    int gtile = blockIdx.x * 8 + wave;                    // 0 .. 4095
    int b = gtile >> 9;                                   // 512 waves per batch
    int t = gtile & 511;
    int c0 = (t >> 5) << 4;                               // channel tile
    int mg = (t & 31) << 6;                               // m group of 64
    const u16* A  = xv   + (size_t)b * CC * NN;
    const u16* Bt = attT + (size_t)b * NN * NN;

    v8f acc[4] = {};
    for (int n0 = 0; n0 < NN; n0 += 32) {
        v16bf a = load_frag_A(A, c0 + l16, NN, n0, half);
        #pragma unroll
        for (int u = 0; u < 4; ++u) {
            v16bf bm = load_frag_B(Bt, mg + u * 16 + l16, NN, n0, half);
            acc[u] = wmma_bf16(a, bm, acc[u]);
        }
    }
    #pragma unroll
    for (int u = 0; u < 4; ++u) {
        int m = mg + u * 16 + l16;
        float inv = 1.0f / (1e-9f + colsum[b * NN + m]);
        #pragma unroll
        for (int v = 0; v < 8; ++v) {
            int c = c0 + v + half * 8;
            float xr = acc[u][v] * inv;
            float y = x[((size_t)b * CC + c) * NN + m] - xr;
            yT[((size_t)b * NN + m) * CC + c] = f2bf(y);
        }
    }
}

// ---------------------------------------------------------------------------
// Kernel 4: trans conv + affine + BN partial stats, 4-wide blocking (j).
// z[b,o,m] = alpha[o]*(sum_c Wt[o,c]*y[b,c,m] + bt[o]) + beta[o]
// BN sum/sumsq accumulated over the 4 j-tiles first, then one wave
// reduction + atomic per channel. 4096 waves; grid=512, block=256.
// ---------------------------------------------------------------------------
__global__ void k4_transconv(const u16* __restrict__ Wt_bf,
                             const u16* __restrict__ yT,    // [BN][C]
                             const float* __restrict__ bt,
                             const float* __restrict__ alpha,
                             const float* __restrict__ beta,
                             float* __restrict__ z,         // [B][C][N]
                             float* __restrict__ bnsum,
                             float* __restrict__ bnsumsq) {
    int lane = threadIdx.x & 31, wave = threadIdx.x >> 5;
    int half = lane >> 4, l16 = lane & 15;
    int gtile = blockIdx.x * 8 + wave;                    // 0 .. 4095
    int o0 = (gtile >> 8) << 4;                           // channel tile
    int jg = (gtile & 255) << 6;                          // (b,m) group of 64

    v8f acc[4] = {};
    #pragma unroll
    for (int kc = 0; kc < 8; ++kc) {
        v16bf a = load_frag_A(Wt_bf, o0 + l16, CC, kc * 32, half);
        #pragma unroll
        for (int u = 0; u < 4; ++u) {
            v16bf bm = load_frag_B(yT, jg + u * 16 + l16, CC, kc * 32, half);
            acc[u] = wmma_bf16(a, bm, acc[u]);
        }
    }
    float zsum[8], zsq[8];
    #pragma unroll
    for (int v = 0; v < 8; ++v) { zsum[v] = 0.0f; zsq[v] = 0.0f; }

    #pragma unroll
    for (int u = 0; u < 4; ++u) {
        int j = jg + u * 16 + l16;
        int b = j >> 11, m = j & (NN - 1);
        #pragma unroll
        for (int v = 0; v < 8; ++v) {
            int o = o0 + v + half * 8;
            float zv = alpha[o] * (acc[u][v] + bt[o]) + beta[o];
            z[((size_t)b * CC + o) * NN + m] = zv;
            zsum[v] += zv;
            zsq[v]  += zv * zv;
        }
    }
    // Reduce across the 16 lanes sharing each channel (same half), then one
    // atomic per channel per wave.
    #pragma unroll
    for (int v = 0; v < 8; ++v) {
        float s = zsum[v], sq = zsq[v];
        #pragma unroll
        for (int off = 8; off >= 1; off >>= 1) {
            s  += __shfl_xor(s,  off);
            sq += __shfl_xor(sq, off);
        }
        if (l16 == 0) {
            int o = o0 + v + half * 8;
            atomicAdd(&bnsum[o], s);
            atomicAdd(&bnsumsq[o], sq);
        }
    }
}

// ---------------------------------------------------------------------------
// Kernel 5: BatchNorm finalize + ReLU + residual. out = x + relu(bn(z)).
// ---------------------------------------------------------------------------
__global__ void k5_final(const float* __restrict__ z,
                         const float* __restrict__ x,
                         const float* __restrict__ bnsum,
                         const float* __restrict__ bnsumsq,
                         const float* __restrict__ gamma,
                         const float* __restrict__ beta_bn,
                         float* __restrict__ out) {
    int idx = blockIdx.x * 256 + threadIdx.x;             // 0 .. B*C*N-1
    int c = (idx / NN) & (CC - 1);
    const float invN = 1.0f / (float)BN;
    float mean = bnsum[c] * invN;
    float var  = bnsumsq[c] * invN - mean * mean;
    float inv  = rsqrtf(var + BN_EPS);
    float v = (z[idx] - mean) * inv;
    v = gamma[c] * v + beta_bn[c];
    out[idx] = x[idx] + fmaxf(v, 0.0f);
}

// ---------------------------------------------------------------------------
// Host launcher
// ---------------------------------------------------------------------------
extern "C" void kernel_launch(void* const* d_in, const int* in_sizes, int n_in,
                              void* d_out, int out_size, void* d_ws, size_t ws_size,
                              hipStream_t stream) {
    const float* x        = (const float*)d_in[0];
    const float* Wq       = (const float*)d_in[1];
    const float* Wk       = (const float*)d_in[2];
    const float* Wv       = (const float*)d_in[3];
    const float* bv       = (const float*)d_in[4];
    const float* Wt       = (const float*)d_in[5];
    const float* bt       = (const float*)d_in[6];
    const float* bn_gamma = (const float*)d_in[7];
    const float* bn_beta  = (const float*)d_in[8];
    const float* alpha    = (const float*)d_in[9];
    const float* beta     = (const float*)d_in[10];
    float* out = (float*)d_out;

    char* ws = (char*)d_ws;
    size_t off = 0;
    auto carve = [&](size_t bytes) {
        char* p = ws + off;
        off += (bytes + 255) & ~(size_t)255;
        return p;
    };
    u16*   xT      = (u16*)  carve((size_t)BB * NN * CC * 2);   // x^T bf16
    u16*   Wq_bf   = (u16*)  carve((size_t)CC * CC * 2);
    u16*   Wk_bf   = (u16*)  carve((size_t)CC * CC * 2);
    u16*   Wv_bf   = (u16*)  carve((size_t)CC * CC * 2);
    u16*   Wt_bf   = (u16*)  carve((size_t)CC * CC * 2);
    u16*   xq      = (u16*)  carve((size_t)BB * NN * CC * 2);   // [B][N][C]
    u16*   xk      = (u16*)  carve((size_t)BB * NN * CC * 2);   // [B][N][C]
    u16*   xv      = (u16*)  carve((size_t)BB * CC * NN * 2);   // [B][C][N]
    u16*   attT    = (u16*)  carve((size_t)BB * NN * NN * 2);   // [B][m][n]
    float* colsum  = (float*)carve((size_t)BB * NN * 4);
    u16*   yT      = (u16*)  carve((size_t)BB * NN * CC * 2);   // [B][N][C]
    float* z       = (float*)carve((size_t)BB * CC * NN * 4);
    float* bnsum   = (float*)carve((size_t)CC * 4);
    float* bnsumsq = (float*)carve((size_t)CC * 4);
    (void)ws_size; (void)in_sizes; (void)n_in; (void)out_size;

    const int elems = BB * CC * NN;                       // 4,194,304

    k0_convert<<<elems / 256, 256, 0, stream>>>(
        x, Wq, Wk, Wv, Wt, xT, Wq_bf, Wk_bf, Wv_bf, Wt_bf,
        colsum, bnsum, bnsumsq);

    k1_proj<<<dim3(512, 3, 1), 256, 0, stream>>>(
        Wq_bf, Wk_bf, Wv_bf, xT, bv, xq, xk, xv);

    k2_energy_softmax<<<BB * (NN / 16), 256, 0, stream>>>(
        xq, xk, attT, colsum);

    k3_apply<<<512, 256, 0, stream>>>(
        xv, attT, colsum, x, yT);

    k4_transconv<<<512, 256, 0, stream>>>(
        Wt_bf, yT, bt, alpha, beta, z, bnsum, bnsumsq);

    k5_final<<<elems / 256, 256, 0, stream>>>(
        z, x, bnsum, bnsumsq, bn_gamma, bn_beta, out);
}
